// VishwamAIModel_67534065762700
// MI455X (gfx1250) — compile-verified, matrix operand order checked by reading, not run on hardware
//
#include <hip/hip_runtime.h>
#include <hip/hip_bf16.h>

// ---------------------------------------------------------------------------
// Model dims
// ---------------------------------------------------------------------------
#define Bv   8
#define Sv   512
#define Dv   512
#define Hv   8
#define Kv   64
#define Fv   2048
#define Ev   4
#define Lv   128
#define Tv   (Bv * Sv)          // 4096 tokens

#define CHUNK_K 128             // K-chunk staged in LDS for GEMM B-panel

typedef __attribute__((ext_vector_type(16))) __bf16 v16bf;
typedef __attribute__((ext_vector_type(8)))  float  v8f;
typedef unsigned int u32;
typedef __attribute__((ext_vector_type(4))) u32 u32x4;
typedef __attribute__((ext_vector_type(8))) int i32x8;
typedef __attribute__((ext_vector_type(4))) int i32x4;

union ABFrag {
    v16bf        v;
    unsigned int u[8];
};

__device__ __forceinline__ unsigned short f2bf(float f) {
    unsigned u = __float_as_uint(f);
    unsigned r = u + 0x7FFFu + ((u >> 16) & 1u);   // round-to-nearest-even
    return (unsigned short)(r >> 16);
}

// 16-bit A/B fragments (16x32): VGPR v of lane-half h holds K-pair base:
__device__ __forceinline__ int kpair(int v, int half) {
    return (v < 4) ? (half * 8 + v * 2) : (16 + half * 8 + (v - 4) * 2);
}

__device__ __forceinline__ float sigmoidf_(float x) {
    return 1.0f / (1.0f + __expf(-x));
}

__device__ __forceinline__ v8f wmma_bf16(const ABFrag& a, const ABFrag& b, v8f c) {
    return __builtin_amdgcn_wmma_f32_16x16x32_bf16(
        false, a.v, false, b.v, (short)0, c, false, false);
}

// ---------------------------------------------------------------------------
// Stage B^T panel chunk (64 rows x CHUNK_K bf16) into LDS.
// Prefer the CDNA5 Tensor Data Mover; fall back to a cooperative copy.
// ---------------------------------------------------------------------------
__device__ __forceinline__ void stage_b_panel(
    unsigned short* ldsB, const unsigned short* BT,
    int n0, int kc, int Kd, int tid, int wave)
{
#if __has_builtin(__builtin_amdgcn_tensor_load_to_lds) && \
    __has_builtin(__builtin_amdgcn_s_wait_tensorcnt)
    if (wave == 0) {
        unsigned long long ga =
            (unsigned long long)(const void*)(BT + (long)n0 * Kd + kc);
        u32 lds_off = (u32)(unsigned long long)(__SIZE_TYPE__)ldsB;

        u32x4 g0;
        g0.x = 1u;                                   // count = 1 valid D#
        g0.y = lds_off;                              // lds_addr (bytes)
        g0.z = (u32)ga;                              // global_addr[31:0]
        g0.w = (u32)((ga >> 32) & 0x1FFFFFFu)        // global_addr[56:32]
             | 0x80000000u;                          // type = 2 ("image")

        i32x8 g1;
        g1[0] = 1 << 16;                             // data_size = 1 -> 2 bytes
        g1[1] = (Kd & 0xFFFF) << 16;                 // tensor_dim0[15:0]
        g1[2] = (Kd >> 16) | (64 << 16);             // tensor_dim0[31:16] | tensor_dim1
        g1[3] = (CHUNK_K << 16);                     // tile_dim0 = 128 (k, contiguous)
        g1[4] = 64;                                  // tile_dim1 = 64 rows
        g1[5] = Kd;                                  // tensor_dim0_stride[31:0]
        g1[6] = 0;
        g1[7] = 0;

        i32x4 gz = {0, 0, 0, 0};
#if __clang_major__ >= 23
        i32x8 gz8 = {0, 0, 0, 0, 0, 0, 0, 0};
        __builtin_amdgcn_tensor_load_to_lds(g0, g1, gz, gz, gz8, 0);
#else
        __builtin_amdgcn_tensor_load_to_lds(g0, g1, gz, gz, 0);
#endif
        __builtin_amdgcn_s_wait_tensorcnt(0);
    }
#else
    for (int i = tid; i < 64 * (CHUNK_K / 2); i += 256) {
        int row = i / (CHUNK_K / 2);
        int cdw = i % (CHUNK_K / 2);
        ((u32*)ldsB)[i] =
            *(const u32*)(BT + (long)(n0 + row) * Kd + kc + cdw * 2);
    }
#endif
}

// ---------------------------------------------------------------------------
// bf16 WMMA GEMM: C[M,N](f32) = A[M,Kd](bf16) * BT[N,Kd](bf16) + bias[N].
// 256-thread block = 8 waves; block owns 256 rows x 64 cols; each wave owns a
// 32x64 tile block (2x4 WMMA tiles). B panel staged in LDS per 128-K chunk
// (TDM when available); A fragments streamed from global.
// ---------------------------------------------------------------------------
__global__ void gemm_bf16_wmma(const unsigned short* __restrict__ A,
                               const unsigned short* __restrict__ BT,
                               const float* __restrict__ bias,
                               float* __restrict__ C,
                               int M, int N, int Kd, int accumulate)
{
    __shared__ unsigned short ldsB[64 * CHUNK_K];   // 16 KB

    const int tid  = threadIdx.x;
    const int lane = tid & 31;
    const int wave = tid >> 5;
    const int l15  = lane & 15;
    const int half = lane >> 4;

    const int colGroups = N >> 6;
    const int cg = blockIdx.x % colGroups;
    const int rb = blockIdx.x / colGroups;
    const int n0 = cg * 64;
    const int m0 = rb * 256 + wave * 32;

    const long arow0 = (long)(m0 + l15) * Kd;
    const long arow1 = (long)(m0 + 16 + l15) * Kd;

    v8f acc[2][4] = {};

    for (int kc = 0; kc < Kd; kc += CHUNK_K) {
        stage_b_panel(ldsB, BT, n0, kc, Kd, tid, wave);
        __syncthreads();

        if (kc + CHUNK_K < Kd) {                     // prefetch next A chunk
            __builtin_prefetch(A + arow0 + kc + CHUNK_K, 0, 0);
            __builtin_prefetch(A + arow1 + kc + CHUNK_K, 0, 0);
        }

        for (int k0 = 0; k0 < CHUNK_K; k0 += 32) {
            ABFrag a0, a1, bfr[4];
#pragma unroll
            for (int v = 0; v < 8; ++v) {
                const int kb = kpair(v, half);
                a0.u[v] = *(const u32*)(A + arow0 + kc + k0 + kb);
                a1.u[v] = *(const u32*)(A + arow1 + kc + k0 + kb);
#pragma unroll
                for (int j = 0; j < 4; ++j)
                    bfr[j].u[v] = *(const u32*)
                        (ldsB + (16 * j + l15) * CHUNK_K + k0 + kb);
            }
#pragma unroll
            for (int j = 0; j < 4; ++j) {
                acc[0][j] = wmma_bf16(a0, bfr[j], acc[0][j]);
                acc[1][j] = wmma_bf16(a1, bfr[j], acc[1][j]);
            }
        }
        __syncthreads();
    }

#pragma unroll
    for (int j = 0; j < 4; ++j) {
        const int col = n0 + 16 * j + l15;
        const float bvv = bias ? bias[col] : 0.0f;
#pragma unroll
        for (int i = 0; i < 2; ++i) {
#pragma unroll
            for (int r = 0; r < 8; ++r) {
                const int row = m0 + 16 * i + r + 8 * half;
                const long o = (long)row * N + col;
                const float val = acc[i][j][r] + bvv;
                if (accumulate) C[o] += val; else C[o] = val;
            }
        }
    }
}

// ---------------------------------------------------------------------------
// Conversions / gathers
// ---------------------------------------------------------------------------
__global__ void cvt_f32_bf16(const float* __restrict__ in,
                             unsigned short* __restrict__ out, int n)
{
    int i = blockIdx.x * blockDim.x + threadIdx.x;
    if (i < n) out[i] = f2bf(in[i]);
}

// W[Kd,N] f32 -> WT[N,Kd] bf16
__global__ void cvt_transpose_w(const float* __restrict__ W,
                                unsigned short* __restrict__ WT,
                                int Kd, int N)
{
    int i = blockIdx.x * blockDim.x + threadIdx.x;
    if (i < Kd * N) {
        int k = i / N, n = i % N;
        WT[(long)n * Kd + k] = f2bf(W[i]);
    }
}

__global__ void gather_embed(const int* __restrict__ ids,
                             const int* __restrict__ route, int e,
                             const float* __restrict__ emb,
                             unsigned short* __restrict__ xb)
{
    int i = blockIdx.x * blockDim.x + threadIdx.x;
    if (i < Tv * Dv) {
        int t = i >> 9, d = i & 511;
        int id = ids[t];
        if (route) id = (route[t] == e) ? id : 0;
        xb[i] = f2bf(emb[(long)id * Dv + d]);
    }
}

// v f32 [T,512] -> vT bf16 [b][h][kk][s]
__global__ void v_transpose(const float* __restrict__ v,
                            unsigned short* __restrict__ vT)
{
    int i = blockIdx.x * blockDim.x + threadIdx.x;
    if (i < Tv * Dv) {
        int t = i >> 9, c = i & 511;
        int b = t >> 9, s = t & 511;
        int h = c >> 6, kk = c & 63;
        vT[(((long)(b * Hv + h) * Kv) + kk) * Sv + s] = f2bf(v[i]);
    }
}

// ---------------------------------------------------------------------------
// Attention: one wave per (b, h, 16-query tile).
// Pass1: scores^T via WMMA (A = K-chunk, B = Q^T). Softmax per lane. Pass2:
// O = P x V via WMMA (A = P from LDS, B = V^T rows, contiguous along s).
// ---------------------------------------------------------------------------
__global__ void attention_wmma(const unsigned short* __restrict__ qb,
                               const unsigned short* __restrict__ kb,
                               const unsigned short* __restrict__ vT,
                               float* __restrict__ out)
{
    __shared__ float          sc[16][Sv];
    __shared__ unsigned short P [16][Sv];

    const int wgid = blockIdx.x;
    const int qt   = wgid & 31;
    const int h    = (wgid >> 5) & 7;
    const int b    = wgid >> 8;
    const int lane = threadIdx.x;
    const int l15  = lane & 15;
    const int half = lane >> 4;
    const int q0   = qt * 16;

    ABFrag bq[2];
#pragma unroll
    for (int kk0 = 0; kk0 < 2; ++kk0)
#pragma unroll
        for (int v = 0; v < 8; ++v) {
            int kbp = kpair(v, half) + kk0 * 32;
            bq[kk0].u[v] = *(const u32*)
                (qb + ((long)b * Sv + q0 + l15) * Dv + h * Kv + kbp);
        }

    for (int s0 = 0; s0 < Sv; s0 += 16) {
        ABFrag ak[2];
#pragma unroll
        for (int kk0 = 0; kk0 < 2; ++kk0)
#pragma unroll
            for (int v = 0; v < 8; ++v) {
                int kbp = kpair(v, half) + kk0 * 32;
                ak[kk0].u[v] = *(const u32*)
                    (kb + ((long)b * Sv + s0 + l15) * Dv + h * Kv + kbp);
            }
        v8f acc = {};
        acc = wmma_bf16(ak[0], bq[0], acc);
        acc = wmma_bf16(ak[1], bq[1], acc);
#pragma unroll
        for (int r = 0; r < 8; ++r)
            sc[l15][s0 + r + 8 * half] = acc[r] * 0.125f;
    }
    __syncthreads();

    if (half == 0) {
        float mx = -1e30f;
        for (int s = 0; s < Sv; ++s) mx = fmaxf(mx, sc[l15][s]);
        float sum = 0.0f;
        for (int s = 0; s < Sv; ++s) {
            float e = __expf(sc[l15][s] - mx);
            sum += e;
            sc[l15][s] = e;
        }
        float inv = 1.0f / sum;
        for (int s = 0; s < Sv; ++s) P[l15][s] = f2bf(sc[l15][s] * inv);
    }
    __syncthreads();

    for (int kt = 0; kt < 4; ++kt) {
        v8f acc = {};
        for (int s0 = 0; s0 < Sv; s0 += 32) {
            ABFrag a, bvf;
#pragma unroll
            for (int v = 0; v < 8; ++v) {
                int kbp = kpair(v, half);
                a.u[v]   = *(const u32*)&P[l15][s0 + kbp];
                bvf.u[v] = *(const u32*)
                    (vT + (((long)(b * Hv + h) * Kv) + kt * 16 + l15) * Sv
                        + s0 + kbp);
            }
            acc = wmma_bf16(a, bvf, acc);
        }
#pragma unroll
        for (int r = 0; r < 8; ++r) {
            int m = q0 + r + 8 * half;
            out[((long)b * Sv + m) * Dv + h * Kv + kt * 16 + l15] = acc[r];
        }
    }
}

// ---------------------------------------------------------------------------
// Gating + argmax (first-max semantics like jnp.argmax)
// ---------------------------------------------------------------------------
__global__ void gating_argmax(const float* __restrict__ hbase,
                              const float* __restrict__ wg,
                              int* __restrict__ route)
{
    int t = blockIdx.x * blockDim.x + threadIdx.x;
    if (t < Tv) {
        float g0 = 0, g1 = 0, g2 = 0, g3 = 0;
        const float* hp = hbase + (long)t * Dv;
        for (int d = 0; d < Dv; ++d) {
            float x = hp[d];
            g0 += x * wg[d * Ev + 0];
            g1 += x * wg[d * Ev + 1];
            g2 += x * wg[d * Ev + 2];
            g3 += x * wg[d * Ev + 3];
        }
        int best = 0; float bvv = g0;
        if (g1 > bvv) { bvv = g1; best = 1; }
        if (g2 > bvv) { bvv = g2; best = 2; }
        if (g3 > bvv) { bvv = g3; best = 3; }
        route[t] = best;
    }
}

// ---------------------------------------------------------------------------
// LSTM scan: one block per batch row, 512 threads (one per z column)
// ---------------------------------------------------------------------------
__global__ void lstm_seq(const float* __restrict__ x,
                         const float* __restrict__ W,
                         const float* __restrict__ bvec,
                         float* __restrict__ outh)
{
    const int b = blockIdx.x;
    const int j = threadIdx.x;
    __shared__ float hC[Lv], cC[Lv], z[4 * Lv];
    if (j < Lv) { hC[j] = 0.0f; cC[j] = 0.0f; }
    __syncthreads();
    for (int s = 0; s < Sv; ++s) {
        const float* xt = x + ((long)b * Sv + s) * Dv;
        float acc = bvec[j];
        for (int d = 0; d < Dv; ++d) acc += xt[d] * W[d * (4 * Lv) + j];
        for (int l = 0; l < Lv; ++l)  acc += hC[l] * W[(Dv + l) * (4 * Lv) + j];
        z[j] = acc;
        __syncthreads();
        if (j < Lv) {
            float i_ = z[j], g_ = z[Lv + j], f_ = z[2 * Lv + j], o_ = z[3 * Lv + j];
            float c2 = sigmoidf_(f_ + 1.0f) * cC[j] + sigmoidf_(i_) * tanhf(g_);
            float h2 = sigmoidf_(o_) * tanhf(c2);
            cC[j] = c2; hC[j] = h2;
            outh[((long)b * Sv + s) * Lv + j] = h2;
        }
        __syncthreads();
    }
}

__global__ void final_proj(const float* __restrict__ lstm,
                           const float* __restrict__ wd,
                           const float* __restrict__ bd,
                           float* __restrict__ out)
{
    int t = blockIdx.x * blockDim.x + threadIdx.x;
    if (t < Tv) {
        float a = bd[0];
        for (int l = 0; l < Lv; ++l) a += lstm[(long)t * Lv + l] * wd[l];
        out[t] = a;
    }
}

// ---------------------------------------------------------------------------
// Host orchestration
// ---------------------------------------------------------------------------
namespace {
constexpr size_t MB = 1024ull * 1024ull;
constexpr size_t OFF_WQT = 0;
constexpr size_t OFF_WKT = 512 * 1024ull;
constexpr size_t OFF_WVT = 1 * MB;
constexpr size_t OFF_WOT = 1 * MB + 512 * 1024ull;
constexpr size_t OFF_W1T = 2 * MB;
constexpr size_t OFF_W2T = 4 * MB;
constexpr size_t OFF_XB  = 6 * MB;
constexpr size_t OFF_YB  = 10 * MB;
constexpr size_t OFF_QB  = 14 * MB;
constexpr size_t OFF_KB  = 18 * MB;
constexpr size_t OFF_VT  = 22 * MB;
constexpr size_t OFF_T1  = 26 * MB;
constexpr size_t OFF_T2  = 34 * MB;
constexpr size_t OFF_H1F = 42 * MB;
constexpr size_t OFF_H1B = 74 * MB;
constexpr size_t OFF_HB  = 90 * MB;
constexpr size_t OFF_AGG = 98 * MB;
constexpr size_t OFF_RT  = 106 * MB;
constexpr size_t OFF_LST = 106 * MB + 64 * 1024ull;

inline int cdiv(long n, int d) { return (int)((n + d - 1) / d); }

void launch_gemm(hipStream_t st, const unsigned short* A, const unsigned short* BT,
                 const float* bias, float* C, int M, int N, int Kd, int acc)
{
    int blocks = (M / 256) * (N / 64);
    gemm_bf16_wmma<<<blocks, 256, 0, st>>>(A, BT, bias, C, M, N, Kd, acc);
}

void launch_cvt(hipStream_t st, const float* in, unsigned short* out, long n)
{
    cvt_f32_bf16<<<cdiv(n, 256), 256, 0, st>>>(in, out, (int)n);
}

void launch_wT(hipStream_t st, const float* W, unsigned short* WT, int Kd, int N)
{
    cvt_transpose_w<<<cdiv((long)Kd * N, 256), 256, 0, st>>>(W, WT, Kd, N);
}

void run_block(hipStream_t st, char* ws,
               const int* ids, const int* route, int e,
               const float* emb, const float* wq, const float* wk,
               const float* wv, const float* wo, const float* w1,
               const float* b1, const float* w2, const float* b2,
               float* out, int accumulate)
{
    unsigned short* wqT = (unsigned short*)(ws + OFF_WQT);
    unsigned short* wkT = (unsigned short*)(ws + OFF_WKT);
    unsigned short* wvT = (unsigned short*)(ws + OFF_WVT);
    unsigned short* woT = (unsigned short*)(ws + OFF_WOT);
    unsigned short* w1T = (unsigned short*)(ws + OFF_W1T);
    unsigned short* w2T = (unsigned short*)(ws + OFF_W2T);
    unsigned short* xb  = (unsigned short*)(ws + OFF_XB);
    unsigned short* yb  = (unsigned short*)(ws + OFF_YB);
    unsigned short* qb  = (unsigned short*)(ws + OFF_QB);
    unsigned short* kbuf= (unsigned short*)(ws + OFF_KB);
    unsigned short* vT  = (unsigned short*)(ws + OFF_VT);
    unsigned short* h1b = (unsigned short*)(ws + OFF_H1B);
    float* t1  = (float*)(ws + OFF_T1);
    float* t2  = (float*)(ws + OFF_T2);
    float* h1f = (float*)(ws + OFF_H1F);

    launch_wT(st, wq, wqT, Dv, Dv);
    launch_wT(st, wk, wkT, Dv, Dv);
    launch_wT(st, wv, wvT, Dv, Dv);
    launch_wT(st, wo, woT, Dv, Dv);
    launch_wT(st, w1, w1T, Dv, Fv);
    launch_wT(st, w2, w2T, Fv, Dv);

    gather_embed<<<cdiv((long)Tv * Dv, 256), 256, 0, st>>>(ids, route, e, emb, xb);

    launch_gemm(st, xb, wqT, nullptr, t1, Tv, Dv, Dv, 0);
    launch_cvt(st, t1, qb, (long)Tv * Dv);
    launch_gemm(st, xb, wkT, nullptr, t1, Tv, Dv, Dv, 0);
    launch_cvt(st, t1, kbuf, (long)Tv * Dv);
    launch_gemm(st, xb, wvT, nullptr, t1, Tv, Dv, Dv, 0);
    v_transpose<<<cdiv((long)Tv * Dv, 256), 256, 0, st>>>(t1, vT);

    attention_wmma<<<Bv * Hv * (Sv / 16), 32, 0, st>>>(qb, kbuf, vT, t1);

    launch_cvt(st, t1, yb, (long)Tv * Dv);
    launch_gemm(st, yb, woT, nullptr, t2, Tv, Dv, Dv, 0);

    launch_cvt(st, t2, xb, (long)Tv * Dv);
    launch_gemm(st, xb, w1T, b1, h1f, Tv, Fv, Dv, 0);
    launch_cvt(st, h1f, h1b, (long)Tv * Fv);
    launch_gemm(st, h1b, w2T, b2, out, Tv, Dv, Fv, accumulate);
}
} // namespace

extern "C" void kernel_launch(void* const* d_in, const int* in_sizes, int n_in,
                              void* d_out, int out_size, void* d_ws, size_t ws_size,
                              hipStream_t stream)
{
    (void)in_sizes; (void)n_in; (void)out_size; (void)ws_size;
    char* ws = (char*)d_ws;

    const int*   ids    = (const int*)  d_in[0];
    const float* embed  = (const float*)d_in[1];
    const float* wq     = (const float*)d_in[2];
    const float* wk     = (const float*)d_in[3];
    const float* wv     = (const float*)d_in[4];
    const float* wo     = (const float*)d_in[5];
    const float* w1     = (const float*)d_in[6];
    const float* b1     = (const float*)d_in[7];
    const float* w2     = (const float*)d_in[8];
    const float* b2     = (const float*)d_in[9];
    const float* wg     = (const float*)d_in[10];
    const float* e_emb  = (const float*)d_in[11];
    const float* e_wq   = (const float*)d_in[12];
    const float* e_wk   = (const float*)d_in[13];
    const float* e_wv   = (const float*)d_in[14];
    const float* e_wo   = (const float*)d_in[15];
    const float* e_w1   = (const float*)d_in[16];
    const float* e_b1   = (const float*)d_in[17];
    const float* e_w2   = (const float*)d_in[18];
    const float* e_b2   = (const float*)d_in[19];
    const float* a_wq   = (const float*)d_in[20];
    const float* a_wk   = (const float*)d_in[21];
    const float* a_wv   = (const float*)d_in[22];
    const float* a_wo   = (const float*)d_in[23];
    const float* lstm_w = (const float*)d_in[24];
    const float* lstm_b = (const float*)d_in[25];
    const float* wd     = (const float*)d_in[26];
    const float* bd     = (const float*)d_in[27];

    float* hbase = (float*)(ws + OFF_HB);
    float* agg   = (float*)(ws + OFF_AGG);
    int*   route = (int*)  (ws + OFF_RT);
    float* lstm  = (float*)(ws + OFF_LST);
    float* t1    = (float*)(ws + OFF_T1);
    float* t2    = (float*)(ws + OFF_T2);
    unsigned short* xb   = (unsigned short*)(ws + OFF_XB);
    unsigned short* yb   = (unsigned short*)(ws + OFF_YB);
    unsigned short* qb   = (unsigned short*)(ws + OFF_QB);
    unsigned short* kbuf = (unsigned short*)(ws + OFF_KB);
    unsigned short* vT   = (unsigned short*)(ws + OFF_VT);
    unsigned short* wqT  = (unsigned short*)(ws + OFF_WQT);
    unsigned short* wkT  = (unsigned short*)(ws + OFF_WKT);
    unsigned short* wvT  = (unsigned short*)(ws + OFF_WVT);
    unsigned short* woT  = (unsigned short*)(ws + OFF_WOT);

    // 1) base block -> hbase
    run_block(stream, ws, ids, nullptr, 0, embed, wq, wk, wv, wo,
              w1, b1, w2, b2, hbase, 0);

    // 2) gating + argmax routing
    gating_argmax<<<cdiv(Tv, 256), 256, 0, stream>>>(hbase, wg, route);

    // 3) experts, accumulated into agg
    hipMemsetAsync(agg, 0, (size_t)Tv * Dv * sizeof(float), stream);
    for (int e = 0; e < Ev; ++e) {
        run_block(stream, ws, ids, route, e,
                  e_emb + (long)e * 50257 * Dv,
                  e_wq + (long)e * Dv * Dv, e_wk + (long)e * Dv * Dv,
                  e_wv + (long)e * Dv * Dv, e_wo + (long)e * Dv * Dv,
                  e_w1 + (long)e * Dv * Fv, e_b1 + (long)e * Fv,
                  e_w2 + (long)e * Fv * Dv, e_b2 + (long)e * Dv,
                  agg, 1);
    }

    // 4) final MHA over agg
    launch_wT(stream, a_wq, wqT, Dv, Dv);
    launch_wT(stream, a_wk, wkT, Dv, Dv);
    launch_wT(stream, a_wv, wvT, Dv, Dv);
    launch_wT(stream, a_wo, woT, Dv, Dv);
    launch_cvt(stream, agg, xb, (long)Tv * Dv);
    launch_gemm(stream, xb, wqT, nullptr, t1, Tv, Dv, Dv, 0);
    launch_cvt(stream, t1, qb, (long)Tv * Dv);
    launch_gemm(stream, xb, wkT, nullptr, t1, Tv, Dv, Dv, 0);
    launch_cvt(stream, t1, kbuf, (long)Tv * Dv);
    launch_gemm(stream, xb, wvT, nullptr, t1, Tv, Dv, Dv, 0);
    v_transpose<<<cdiv((long)Tv * Dv, 256), 256, 0, stream>>>(t1, vT);
    attention_wmma<<<Bv * Hv * (Sv / 16), 32, 0, stream>>>(qb, kbuf, vT, t1);
    launch_cvt(stream, t1, yb, (long)Tv * Dv);
    launch_gemm(stream, yb, woT, nullptr, t2, Tv, Dv, Dv, 0);

    // 5) LSTM over sequence + final projection
    lstm_seq<<<Bv, 4 * Lv, 0, stream>>>(t2, lstm_w, lstm_b, lstm);
    final_proj<<<cdiv(Tv, 256), 256, 0, stream>>>(lstm, wd, bd, (float*)d_out);
}
